// GaussianAttrMLP_57552561766779
// MI455X (gfx1250) — matrix operand order repaired and checked by reading, not compiled, not run on hardware
//
#include <hip/hip_runtime.h>
#include <math.h>

typedef __attribute__((ext_vector_type(16))) _Float16 v16h;
typedef __attribute__((ext_vector_type(8)))  _Float16 v8h;
typedef __attribute__((ext_vector_type(8)))  float    v8f;

#define HDIM  512
#define ZDIM  256
#define NPTS  32768
#define PITCH 520            // 512 + 8 halfs pad (bank-conflict avoidance)
#define MTILE 64
#define K0    96             // fc1 (64) + fcma (9) padded to 96

// ---------------------------------------------------------------------------
// Pre-kernel 1: per-batch modulation vectors
//   modA[L][b][h] = z[b] . Wa[L][h] + ba[L][h]   (alpha, input scale)
//   modB[L][b][h] = z[b] . Wb[L][h] + bb[L][h]   (beta,  output bias)
//   L = 0..3 trunk, 4..7 heads (xyz,rgb,scale,opacity)
// ---------------------------------------------------------------------------
__global__ void mod_kernel(const float* __restrict__ z,
                           const float* __restrict__ tWa, const float* __restrict__ tba,
                           const float* __restrict__ tWb, const float* __restrict__ tbb,
                           const float* __restrict__ hWa, const float* __restrict__ hba,
                           const float* __restrict__ hWb, const float* __restrict__ hbb,
                           float* __restrict__ modA, float* __restrict__ modB) {
  int idx = blockIdx.x * blockDim.x + threadIdx.x;
  if (idx >= 8 * 4 * HDIM) return;
  int h = idx % HDIM;
  int b = (idx / HDIM) & 3;
  int L = idx / (4 * HDIM);
  const float *Wa, *ba, *Wb, *bb;
  if (L < 4) {
    Wa = tWa + ((size_t)L * HDIM + h) * ZDIM;  ba = tba + L * HDIM + h;
    Wb = tWb + ((size_t)L * HDIM + h) * ZDIM;  bb = tbb + L * HDIM + h;
  } else {
    int l = L - 4;
    Wa = hWa + ((size_t)l * HDIM + h) * ZDIM;  ba = hba + l * HDIM + h;
    Wb = hWb + ((size_t)l * HDIM + h) * ZDIM;  bb = hbb + l * HDIM + h;
  }
  const float* zb = z + b * ZDIM;
  float sa = 0.f, sb = 0.f;
  for (int k = 0; k < ZDIM; ++k) { sa += zb[k] * Wa[k]; sb += zb[k] * Wb[k]; }
  modA[idx] = sa + *ba;
  modB[idx] = sb + *bb;
}

// ---------------------------------------------------------------------------
// Pre-kernel 2: repack 4x trunk [512,512] f32 weights into f16 WMMA-B order
//   layout: [L][cb(32)][kb(16)][lane(32)][j(16)]
//   element = W[L][ cb*16 + lane%16 ][ kb*32 + (lane/16)*16 + j ]
//   -> each WMMA B-fragment load = 32 contiguous bytes per lane, coalesced
// ---------------------------------------------------------------------------
__global__ void pack_w_kernel(const float* __restrict__ tW,
                              _Float16* __restrict__ Wp) {
  int idx = blockIdx.x * blockDim.x + threadIdx.x;   // 4*512*512 = 1048576
  if (idx >= 4 * HDIM * HDIM) return;
  int j    = idx & 15;
  int lane = (idx >> 4) & 31;
  int kb   = (idx >> 9) & 15;
  int cb   = (idx >> 13) & 31;
  int L    = idx >> 18;
  int n = cb * 16 + (lane & 15);
  int k = kb * 32 + (lane >> 4) * 16 + j;
  Wp[idx] = (_Float16)tW[((size_t)L * HDIM + n) * HDIM + k];
}

// ---------------------------------------------------------------------------
// Pre-kernel 3: per-batch head weights with alpha folded in:
//   WpH[h][b][...] = head_W[h][n][k] * modA[4+h][b][k]
//   ((x*alpha) @ W^T == x @ (W*diag(alpha))^T, alpha is per-(batch,k))
// ---------------------------------------------------------------------------
__global__ void pack_wh_kernel(const float* __restrict__ hW,
                               const float* __restrict__ modA,
                               _Float16* __restrict__ WpH) {
  int idx = blockIdx.x * blockDim.x + threadIdx.x;   // 4*4*512*512 = 4194304
  if (idx >= 4 * 4 * HDIM * HDIM) return;
  int j    = idx & 15;
  int lane = (idx >> 4) & 31;
  int kb   = (idx >> 9) & 15;
  int cb   = (idx >> 13) & 31;
  int b    = (idx >> 18) & 3;
  int h    = idx >> 20;
  int n = cb * 16 + (lane & 15);
  int k = kb * 32 + (lane >> 4) * 16 + j;
  float a = modA[(size_t)(4 + h) * 4 * HDIM + (size_t)b * HDIM + k];
  WpH[idx] = (_Float16)(hW[((size_t)h * HDIM + n) * HDIM + k] * a);
}

// Pre-kernel 4: combined layer-0 weight [512, 96] (fc1 | fcma | zero-pad)
__global__ void pack_w0_kernel(const float* __restrict__ fc1_w,
                               const float* __restrict__ fcma_w,
                               _Float16* __restrict__ W0p) {
  int idx = blockIdx.x * blockDim.x + threadIdx.x;   // 32*3*32*16 = 49152
  if (idx >= HDIM * K0) return;
  int j    = idx & 15;
  int lane = (idx >> 4) & 31;
  int kb   = (idx >> 9) % 3;
  int cb   = idx / (3 * 512);
  int n = cb * 16 + (lane & 15);
  int k = kb * 32 + (lane >> 4) * 16 + j;
  float v = 0.f;
  if (k < 64)      v = fc1_w[n * 64 + k];
  else if (k < 73) v = fcma_w[n * 9 + (k - 64)];
  W0p[idx] = (_Float16)v;
}

// ---------------------------------------------------------------------------
// A-fragment loader (contiguous k order k = 16*g + s, same as the packed B fragments)
// A 16x32 f16: lane group g=lane/16, row = lane%16,
//   halfs 0..7  = K[kbase + g*16 .. +7],  halfs 8..15 = K[kbase+g*16+8 .. +7]
// ---------------------------------------------------------------------------
__device__ __forceinline__ v16h load_a_frag(const _Float16* __restrict__ aBase,
                                            int rowOff, int kbase) {
  const _Float16* p = aBase + rowOff * PITCH + kbase;
  v8h lo = *(const v8h*)p;
  v8h hi = *(const v8h*)(p + 8);
  return __builtin_shufflevector(lo, hi, 0,1,2,3,4,5,6,7,8,9,10,11,12,13,14,15);
}

// ---------------------------------------------------------------------------
// One fused layer: Y[64,512] = LReLU( X[64,Kb*32] @ W^T + bias ) [* scaleOut]
// per wave: 64-column stripe; 4 row-tiles x 4 col-tiles x Kb k-steps WMMA.
// Kb stays a RUNTIME argument: the compiler's bounded partial unroll (~3 ksteps)
// keeps B fragments register-resident across all row-tiles without duplicating
// loads.
// HAS_SCALE is compile-time -> branch-free epilogue.
// ---------------------------------------------------------------------------
template <bool HAS_SCALE>
__device__ __forceinline__ void gemm_layer(const _Float16* __restrict__ X,
                                           _Float16* __restrict__ Y,
                                           const _Float16* __restrict__ Wlay,
                                           int Kb,
                                           const float* __restrict__ bias,
                                           const float* __restrict__ scaleOut,
                                           int lane, int wave) {
  const int r = lane & 15;
  const int g = lane >> 4;
  const _Float16* aBase = X + r * PITCH + g * 16;
  const _Float16* wBase = Wlay + ((size_t)wave * 4 * Kb * 32 + lane) * 16;
  _Float16* yBase = Y + (g * 8) * PITCH + wave * 64 + r;

  // bias / store-scale hoisted out of the row-tile loop (4 loads per layer)
  float bv[4], sc[4];
#pragma unroll
  for (int c = 0; c < 4; ++c) {
    int n = wave * 64 + c * 16 + r;
    bv[c] = bias[n];
    if (HAS_SCALE) sc[c] = scaleOut[n];
  }

  for (int rt = 0; rt < 4; ++rt) {
    v8f acc[4] = {};
    for (int kb = 0; kb < Kb; ++kb) {
      v16h a = load_a_frag(aBase, rt * 16, kb * 32);
#pragma unroll
      for (int c = 0; c < 4; ++c) {
        v16h bf = *(const v16h*)(wBase + (size_t)(c * Kb + kb) * 32 * 16);
        acc[c] = __builtin_amdgcn_wmma_f32_16x16x32_f16(
            false, a, false, bf, (short)0, acc[c], false, false);
      }
    }
    // epilogue: + beta/bias, LeakyReLU(0.2), [* alpha_of_next_layer], -> f16 LDS
#pragma unroll
    for (int c = 0; c < 4; ++c) {
#pragma unroll
      for (int r8 = 0; r8 < 8; ++r8) {
        float v = acc[c][r8] + bv[c];
        v = (v > 0.f) ? v : 0.2f * v;
        if (HAS_SCALE) v *= sc[c];
        yBase[(rt * 16 + r8) * PITCH + c * 16] = (_Float16)v;
      }
    }
  }
}

// ---------------------------------------------------------------------------
// Head output projection: out[r, base..base+CNT) = nl( G[r,:] @ ow^T + ob )
// 256 threads = 64 rows x 4 split-K parts of 128; LDS tree reduce.
// MODE: 0 = sigmoid(x)-0.5 (xyz), 1 = identity (rgb), 2 = sigmoid (opacity)
// ---------------------------------------------------------------------------
template <int CNT, int MODE>
__device__ __forceinline__ void project_head(const _Float16* __restrict__ G,
                                             float* __restrict__ proj,
                                             const float* __restrict__ ow,
                                             const float* __restrict__ ob,
                                             float* __restrict__ out,
                                             int row0, int base, int tid) {
  int r = tid >> 2, p = tid & 3;
  float s[3] = {0.f, 0.f, 0.f};
  const _Float16* g = G + (size_t)r * PITCH + p * 128;
  const float* w = ow + p * 128;
  for (int k = 0; k < 128; ++k) {
    float gv = (float)g[k];
#pragma unroll
    for (int c = 0; c < CNT; ++c) s[c] += gv * w[c * HDIM + k];
  }
#pragma unroll
  for (int c = 0; c < CNT; ++c) proj[((r << 2) + p) * 3 + c] = s[c];
  __syncthreads();
  if (p == 0) {
#pragma unroll
    for (int c = 0; c < CNT; ++c) {
      float v = proj[(r << 2) * 3 + c] + proj[((r << 2) + 1) * 3 + c] +
                proj[((r << 2) + 2) * 3 + c] + proj[((r << 2) + 3) * 3 + c] + ob[c];
      if (MODE == 0)      v = 1.f / (1.f + __expf(-v)) - 0.5f;
      else if (MODE == 2) v = 1.f / (1.f + __expf(-v));
      out[(size_t)(row0 + r) * 10 + base + c] = v;
    }
  }
  __syncthreads();
}

// ---------------------------------------------------------------------------
// Fused persistent-tile MLP: 64 points/block, all 8 layers in LDS (ping-pong)
// ---------------------------------------------------------------------------
__global__ __launch_bounds__(256) void mlp_main(
    const float* __restrict__ pt, const float* __restrict__ oh,
    const _Float16* __restrict__ Wp, const _Float16* __restrict__ WpH,
    const _Float16* __restrict__ W0p,
    const float* __restrict__ modA, const float* __restrict__ modB,
    const float* __restrict__ fc1_b,
    const float* __restrict__ ow_xyz, const float* __restrict__ ob_xyz,
    const float* __restrict__ ow_rgb, const float* __restrict__ ob_rgb,
    const float* __restrict__ ow_op,  const float* __restrict__ ob_op,
    float* __restrict__ out) {
  extern __shared__ _Float16 smem[];
  _Float16* bufA = smem;
  _Float16* bufB = smem + (size_t)MTILE * PITCH;
  float* proj = (float*)(smem + (size_t)2 * MTILE * PITCH);

  int tid = threadIdx.x, lane = tid & 31, wave = tid >> 5;
  int row0 = blockIdx.x * MTILE;
  int b = row0 / NPTS;                       // one batch per tile (N % 64 == 0)

  // trunk alpha (store-scale) / beta (bias) slices for this batch
  const float* A0 = modA + (size_t)(0 * 4 + b) * HDIM;
  const float* A1 = modA + (size_t)(1 * 4 + b) * HDIM;
  const float* A2 = modA + (size_t)(2 * 4 + b) * HDIM;
  const float* A3 = modA + (size_t)(3 * 4 + b) * HDIM;
  const float* B0 = modB + (size_t)(0 * 4 + b) * HDIM;
  const float* B1 = modB + (size_t)(1 * 4 + b) * HDIM;
  const float* B2 = modB + (size_t)(2 * 4 + b) * HDIM;
  const float* B3 = modB + (size_t)(3 * 4 + b) * HDIM;

  // stage layer-0 input [64][96] = [pt | onehot | 0] as f16 into bufA
  for (int idx = tid; idx < MTILE * K0; idx += 256) {
    int r = idx / K0, k = idx - r * K0;
    float v = 0.f;
    if (k < 64)      v = pt[(size_t)(row0 + r) * 64 + k];
    else if (k < 73) v = oh[(size_t)(row0 + r) * 9 + (k - 64)];
    bufA[(size_t)r * PITCH + k] = (_Float16)v;
  }
  __syncthreads();

  // layer0: Kb=3, bias=fc1_b, store * alpha(trunk0)
  gemm_layer<true>(bufA, bufB, W0p, 3, fc1_b, A0, lane, wave);
  __syncthreads();
  // trunk chain (alpha pre-folded at previous store; beta as bias here)
  gemm_layer<true>(bufB, bufA, Wp + (size_t)0 * HDIM * HDIM, 16, B0, A1, lane, wave);
  __syncthreads();
  gemm_layer<true>(bufA, bufB, Wp + (size_t)1 * HDIM * HDIM, 16, B1, A2, lane, wave);
  __syncthreads();
  gemm_layer<true>(bufB, bufA, Wp + (size_t)2 * HDIM * HDIM, 16, B2, A3, lane, wave);
  __syncthreads();
  gemm_layer<false>(bufA, bufB, Wp + (size_t)3 * HDIM * HDIM, 16, B3, nullptr, lane, wave);
  __syncthreads();
  // bufB holds raw trunk output f; head alpha is folded into per-batch weights

  // head xyz: channels 0..2, sigmoid-0.5
  {
    const _Float16* Wh = WpH + (size_t)(0 * 4 + b) * HDIM * HDIM;
    const float* Bh = modB + (size_t)(4 * 4 + b) * HDIM;
    gemm_layer<false>(bufB, bufA, Wh, 16, Bh, nullptr, lane, wave);
    __syncthreads();
    project_head<3, 0>(bufA, proj, ow_xyz, ob_xyz, out, row0, 0, tid);
  }
  // head rgb: channels 3..5, identity
  {
    const _Float16* Wh = WpH + (size_t)(1 * 4 + b) * HDIM * HDIM;
    const float* Bh = modB + (size_t)(5 * 4 + b) * HDIM;
    gemm_layer<false>(bufB, bufA, Wh, 16, Bh, nullptr, lane, wave);
    __syncthreads();
    project_head<3, 1>(bufA, proj, ow_rgb, ob_rgb, out, row0, 3, tid);
  }
  // head scale: clip(x,1,1) == 1.0 -> skipped, constant written below
  // head opacity: channel 9, sigmoid
  {
    const _Float16* Wh = WpH + (size_t)(3 * 4 + b) * HDIM * HDIM;
    const float* Bh = modB + (size_t)(7 * 4 + b) * HDIM;
    gemm_layer<false>(bufB, bufA, Wh, 16, Bh, nullptr, lane, wave);
    __syncthreads();
    project_head<1, 2>(bufA, proj, ow_op, ob_op, out, row0, 9, tid);
  }
  // constant scale channels 6..8
  if (tid < 192) {
    int r = tid / 3, c = tid % 3;
    out[(size_t)(row0 + r) * 10 + 6 + c] = 1.0f;
  }
}

// ---------------------------------------------------------------------------
extern "C" void kernel_launch(void* const* d_in, const int* in_sizes, int n_in,
                              void* d_out, int out_size, void* d_ws, size_t ws_size,
                              hipStream_t stream) {
  (void)in_sizes; (void)n_in; (void)out_size; (void)ws_size;
  const float* pt     = (const float*)d_in[0];
  const float* oh     = (const float*)d_in[1];
  const float* z      = (const float*)d_in[2];
  const float* fc1_w  = (const float*)d_in[3];
  const float* fc1_b  = (const float*)d_in[4];
  const float* fcma_w = (const float*)d_in[5];
  const float* tW  = (const float*)d_in[6];
  const float* tWa = (const float*)d_in[7];
  const float* tba = (const float*)d_in[8];
  const float* tWb = (const float*)d_in[9];
  const float* tbb = (const float*)d_in[10];
  const float* hW  = (const float*)d_in[11];
  const float* hWa = (const float*)d_in[12];
  const float* hba = (const float*)d_in[13];
  const float* hWb = (const float*)d_in[14];
  const float* hbb = (const float*)d_in[15];
  const float* ow_xyz = (const float*)d_in[16];
  const float* ob_xyz = (const float*)d_in[17];
  const float* ow_rgb = (const float*)d_in[18];
  const float* ob_rgb = (const float*)d_in[19];
  const float* ow_op  = (const float*)d_in[22];
  const float* ob_op  = (const float*)d_in[23];
  float* out = (float*)d_out;

  // workspace layout
  char* ws = (char*)d_ws;
  size_t off = 0;
  _Float16* Wp  = (_Float16*)(ws + off); off += (size_t)4 * HDIM * HDIM * 2;      // 2 MB trunk
  _Float16* WpH = (_Float16*)(ws + off); off += (size_t)4 * 4 * HDIM * HDIM * 2;  // 8 MB heads*batch
  _Float16* W0p = (_Float16*)(ws + off); off += (size_t)HDIM * K0 * 2;            // layer0
  float* modA = (float*)(ws + off); off += (size_t)8 * 4 * HDIM * 4;
  float* modB = (float*)(ws + off);

  mod_kernel<<<64, 256, 0, stream>>>(z, tWa, tba, tWb, tbb, hWa, hba, hWb, hbb, modA, modB);
  pack_w_kernel<<<4096, 256, 0, stream>>>(tW, Wp);
  pack_wh_kernel<<<16384, 256, 0, stream>>>(hW, modA, WpH);   // after mod_kernel (same stream)
  pack_w0_kernel<<<192, 256, 0, stream>>>(fc1_w, fcma_w, W0p);

  size_t smem = (size_t)2 * MTILE * PITCH * sizeof(_Float16) + 64 * 4 * 3 * sizeof(float);
  (void)hipFuncSetAttribute((const void*)mlp_main,
                            hipFuncAttributeMaxDynamicSharedMemorySize, (int)smem);
  int nTiles = (4 * NPTS) / MTILE;  // 2048
  mlp_main<<<nTiles, 256, smem, stream>>>(pt, oh, Wp, WpH, W0p, modA, modB, fc1_b,
                                          ow_xyz, ob_xyz, ow_rgb, ob_rgb,
                                          ow_op, ob_op, out);
}
